// CrossMamba_40415642255958
// MI455X (gfx1250) — compile-verified, matrix operand order checked by reading, not run on hardware
//
#include <hip/hip_runtime.h>
#include <hip/hip_bf16.h>

#define EMBED  768
#define DEPTH  4
#define BATCH  4
#define SEQ    1024
#define DIN    1536
#define DSTATE 16
#define DCONV  4
#define DTRANK 48
#define BL     (BATCH * SEQ)          /* 4096 rows */
#define NPROJ  (DTRANK + 2 * DSTATE)  /* 80 */

// Flip to 0 if the assembler rejects the async-to-LDS mnemonics.
#define USE_ASYNC 1

typedef __attribute__((ext_vector_type(16))) __bf16 v16bf;
typedef __attribute__((ext_vector_type(8)))  __bf16 v8bf;
typedef __attribute__((ext_vector_type(8)))  float  v8f;

__device__ __forceinline__ float silu_f(float x) { return x / (1.f + __expf(-x)); }

__device__ __forceinline__ v16bf frag16(const __bf16* p)
{
    return __builtin_shufflevector(*(const v8bf*)p, *(const v8bf*)(p + 16),
                                   0, 1, 2, 3, 4, 5, 6, 7, 8, 9, 10, 11, 12, 13, 14, 15);
}

__device__ __forceinline__ v8f wmma_bf16(v16bf a, v16bf b, v8f c)
{
    return __builtin_amdgcn_wmma_f32_16x16x32_bf16(false, a, false, b, (short)0, c, false, false);
}

// 16-byte global -> LDS transfer (async on CDNA5; sync fallback via VGPR)
__device__ __forceinline__ void ld16_to_lds(__bf16* lds_p, const __bf16* gp)
{
#if USE_ASYNC
    unsigned la = (unsigned)(unsigned long long)(uintptr_t)lds_p;
    asm volatile("global_load_async_to_lds_b128 %0, %1, off"
                 :: "v"(la), "v"(gp) : "memory");
#else
    *(v8bf*)lds_p = *(const v8bf*)gp;
#endif
}

__device__ __forceinline__ void wait_lds_fill()
{
#if USE_ASYNC
    asm volatile("s_wait_asynccnt 0x0" ::: "memory");
#endif
}

// ---------------------------------------------------------------------------
// fp32 -> bf16 convert with optional column pad (dst cols >= src cols)
// ---------------------------------------------------------------------------
__global__ void cvt_pad_kernel(const float* __restrict__ src, __bf16* __restrict__ dst,
                               int rows, int sstride, int cin, int cout)
{
    int idx = blockIdx.x * 256 + threadIdx.x;
    int total = rows * cout;
    if (idx >= total) return;
    int r = idx / cout;
    int c = idx - r * cout;
    float v = (c < cin) ? src[(long)r * sstride + c] : 0.f;
    dst[idx] = (__bf16)v;
}

// ---------------------------------------------------------------------------
// residual += hidden; hn = rmsnorm(residual) * w  (bf16 out). one block per row
// ---------------------------------------------------------------------------
__global__ void rmsnorm_kernel(const float* __restrict__ h_in, float* __restrict__ resid,
                               const float* __restrict__ w, __bf16* __restrict__ hn, int first)
{
    int row = blockIdx.x;
    int t = threadIdx.x;
    int base = row * EMBED;
    float v[3];
    float ss = 0.f;
#pragma unroll
    for (int j = 0; j < 3; ++j) {
        int c = t + j * 256;
        float x = h_in[base + c];
        if (!first) x += resid[base + c];
        resid[base + c] = x;
        v[j] = x;
        ss += x * x;
    }
    __shared__ float red[256];
    red[t] = ss;
    __syncthreads();
    for (int s = 128; s > 0; s >>= 1) {
        if (t < s) red[t] += red[t + s];
        __syncthreads();
    }
    float scale = rsqrtf(red[0] * (1.f / EMBED) + 1e-5f);
#pragma unroll
    for (int j = 0; j < 3; ++j) {
        int c = t + j * 256;
        hn[base + c] = (__bf16)(v[j] * scale * w[c]);
    }
}

// ---------------------------------------------------------------------------
// out = rmsnorm(hidden + residual) * w + b  (fp32 out). one block per row
// ---------------------------------------------------------------------------
__global__ void final_norm_kernel(const float* __restrict__ hidden, const float* __restrict__ resid,
                                  const float* __restrict__ w, const float* __restrict__ bia,
                                  float* __restrict__ out)
{
    int row = blockIdx.x;
    int t = threadIdx.x;
    int base = row * EMBED;
    float v[3];
    float ss = 0.f;
#pragma unroll
    for (int j = 0; j < 3; ++j) {
        int c = t + j * 256;
        float x = hidden[base + c] + resid[base + c];
        v[j] = x;
        ss += x * x;
    }
    __shared__ float red[256];
    red[t] = ss;
    __syncthreads();
    for (int s = 128; s > 0; s >>= 1) {
        if (t < s) red[t] += red[t + s];
        __syncthreads();
    }
    float scale = rsqrtf(red[0] * (1.f / EMBED) + 1e-5f);
#pragma unroll
    for (int j = 0; j < 3; ++j) {
        int c = t + j * 256;
        out[base + c] = v[j] * scale * w[c] + bia[c];
    }
}

// ---------------------------------------------------------------------------
// NT GEMM: C[M][N] = A[M][K] * B[N][K]^T, bf16 in, fp32 WMMA accumulate.
// block = 256 threads / 8 waves in 2x4; block tile 64(M) x 128(N);
// each wave: 32x32 output (4 accumulators), K-step 64 (8 WMMA per step).
// Double-buffered LDS, async global->LDS fills, one barrier per K-step.
// K must be a multiple of 64; M = gridDim.y*64 exact; N guarded (OOB rows
// clamped on load, columns guarded on store).
// EPI 0: store fp32.  EPI 1: store bf16 of silu(x).
// ---------------------------------------------------------------------------
#define KSTEP 64
#define LDSS  72  /* LDS row stride (bf16 elems): conflict-free fragment reads */

template <int EPI>
__global__ void gemm_nt_bf16(const __bf16* __restrict__ A, const __bf16* __restrict__ Bw,
                             void* __restrict__ Cout, int N, int K)
{
    __shared__ __align__(16) __bf16 As[2][64 * LDSS];
    __shared__ __align__(16) __bf16 Bs[2][128 * LDSS];

    int tid  = threadIdx.x;
    int wave = tid >> 5;
    int lane = tid & 31;
    int wm   = wave >> 2;   // 0..1 : 32-row slab
    int wn   = wave & 3;    // 0..3 : 32-col slab
    int g    = lane >> 4;
    int r    = lane & 15;

    int bm = blockIdx.y * 64;
    int bn = blockIdx.x * 128;

    // tile fill: A 64x64 (512 x 16B chunks), B 128x64 (1024 x 16B chunks)
    auto load_tile = [&](int buf, int k0) {
#pragma unroll
        for (int q = 0; q < 2; ++q) {
            int chunk = tid + 256 * q;
            int row = chunk >> 3, cb = chunk & 7;
            ld16_to_lds(&As[buf][row * LDSS + cb * 8],
                        A + (long)(bm + row) * K + k0 + cb * 8);
        }
#pragma unroll
        for (int q = 0; q < 4; ++q) {
            int chunk = tid + 256 * q;
            int row = chunk >> 3, cb = chunk & 7;
            int n = bn + row;
            if (n > N - 1) n = N - 1;  // clamp: never touch OOB memory
            ld16_to_lds(&Bs[buf][row * LDSS + cb * 8],
                        Bw + (long)n * K + k0 + cb * 8);
        }
    };

    v8f acc00 = {}, acc01 = {}, acc10 = {}, acc11 = {};

    load_tile(0, 0);
    wait_lds_fill();
    __syncthreads();

    int nk = K / KSTEP;
    for (int kk = 0; kk < nk; ++kk) {
        int buf = kk & 1;
        if (kk + 1 < nk) load_tile(buf ^ 1, (kk + 1) * KSTEP);

#pragma unroll
        for (int s = 0; s < 2; ++s) {
            const __bf16* ap0 = &As[buf][(wm * 32 + r) * LDSS + s * 32 + g * 8];
            const __bf16* bp0 = &Bs[buf][(wn * 32 + r) * LDSS + s * 32 + g * 8];
            v16bf a0 = frag16(ap0);
            v16bf a1 = frag16(ap0 + 16 * LDSS);
            v16bf b0 = frag16(bp0);
            v16bf b1 = frag16(bp0 + 16 * LDSS);
            acc00 = wmma_bf16(a0, b0, acc00);
            acc01 = wmma_bf16(a0, b1, acc01);
            acc10 = wmma_bf16(a1, b0, acc10);
            acc11 = wmma_bf16(a1, b1, acc11);
        }

        if (kk + 1 < nk) wait_lds_fill();
        __syncthreads();
    }

    int row0 = bm + wm * 32 + g * 8;   // acc VGPR j -> row row0+j (+16 for a1 tiles)
    int col0 = bn + wn * 32 + r;       // b0 col; b1 col = col0+16
    if (EPI == 0) {
        float* C = (float*)Cout;
#pragma unroll
        for (int j = 0; j < 8; ++j) {
            if (col0 < N) {
                C[(long)(row0 + j) * N + col0]      = acc00[j];
                C[(long)(row0 + 16 + j) * N + col0] = acc10[j];
            }
            if (col0 + 16 < N) {
                C[(long)(row0 + j) * N + col0 + 16]      = acc01[j];
                C[(long)(row0 + 16 + j) * N + col0 + 16] = acc11[j];
            }
        }
    } else {
        __bf16* C = (__bf16*)Cout;
#pragma unroll
        for (int j = 0; j < 8; ++j) {
            if (col0 < N) {
                C[(long)(row0 + j) * N + col0]      = (__bf16)silu_f(acc00[j]);
                C[(long)(row0 + 16 + j) * N + col0] = (__bf16)silu_f(acc10[j]);
            }
            if (col0 + 16 < N) {
                C[(long)(row0 + j) * N + col0 + 16]      = (__bf16)silu_f(acc01[j]);
                C[(long)(row0 + 16 + j) * N + col0 + 16] = (__bf16)silu_f(acc11[j]);
            }
        }
    }
}

// ---------------------------------------------------------------------------
// u = silu(causal_conv1d(x, w, b)); x = xz[:, :DIN]
// ---------------------------------------------------------------------------
__global__ void conv_silu_kernel(const float* __restrict__ xz, const float* __restrict__ cw,
                                 const float* __restrict__ cb, float* __restrict__ u)
{
    int idx = blockIdx.x * 256 + threadIdx.x;  // BL*DIN exact
    int d  = idx % DIN;
    int bl = idx / DIN;
    int l  = bl & (SEQ - 1);
    float acc = cb[d];
#pragma unroll
    for (int k = 0; k < DCONV; ++k) {
        int ls = l + k - (DCONV - 1);
        if (ls >= 0)
            acc += xz[(long)(bl + k - (DCONV - 1)) * (2 * DIN) + d] * cw[d * DCONV + k];
    }
    u[idx] = silu_f(acc);
}

// ---------------------------------------------------------------------------
// Selective scan: thread = (batch, channel). 16-wide state in registers.
// B/C rows staged through LDS in 8-step chunks. Fuses softplus(dt)+bias,
// D-skip, silu(z) gate; writes ymul as bf16 for the output GEMM.
// ---------------------------------------------------------------------------
__global__ void scan_kernel(const float* __restrict__ dbl, const float* __restrict__ dtpre,
                            const float* __restrict__ u, const float* __restrict__ xz,
                            const float* __restrict__ A_log, const float* __restrict__ b_dt,
                            const float* __restrict__ D_skip, __bf16* __restrict__ ymul)
{
    int b = blockIdx.y;
    int d = blockIdx.x * 256 + threadIdx.x;

    float A[DSTATE];
#pragma unroll
    for (int n = 0; n < DSTATE; ++n) A[n] = -__expf(A_log[d * DSTATE + n]);
    float bdt = b_dt[d];
    float Dsk = D_skip[d];

    float h[DSTATE];
#pragma unroll
    for (int n = 0; n < DSTATE; ++n) h[n] = 0.f;

    __shared__ float bc[8 * 32];  // 8 timesteps x (B[16] | C[16])
    int lrow = threadIdx.x >> 5;
    int col  = threadIdx.x & 31;

    for (int lc = 0; lc < SEQ / 8; ++lc) {
        __syncthreads();
        bc[threadIdx.x] = dbl[(long)((b << 10) + (lc << 3) + lrow) * NPROJ + DTRANK + col];
        __syncthreads();
#pragma unroll
        for (int j = 0; j < 8; ++j) {
            int bl = (b << 10) + (lc << 3) + j;
            float xdt = dtpre[(long)bl * DIN + d] + bdt;
            float dt  = (xdt > 20.f) ? xdt : log1pf(__expf(xdt));
            float ut  = u[(long)bl * DIN + d];
            float dtu = dt * ut;
            const float* Brow = &bc[j * 32];
            float y = 0.f;
#pragma unroll
            for (int n = 0; n < DSTATE; ++n) {
                h[n] = __expf(dt * A[n]) * h[n] + dtu * Brow[n];
                y += h[n] * Brow[16 + n];
            }
            y += ut * Dsk;
            float z = xz[(long)bl * (2 * DIN) + DIN + d];
            ymul[(long)bl * DIN + d] = (__bf16)(y * silu_f(z));
        }
    }
}

// ---------------------------------------------------------------------------
extern "C" void kernel_launch(void* const* d_in, const int* in_sizes, int n_in,
                              void* d_out, int out_size, void* d_ws, size_t ws_size,
                              hipStream_t stream)
{
    const float* x_mamba = (const float*)d_in[0];
    const float* x_attn  = (const float*)d_in[1];
    const float* W_in    = (const float*)d_in[2];
    const float* W_extra = (const float*)d_in[3];
    const float* conv_w  = (const float*)d_in[4];
    const float* conv_b  = (const float*)d_in[5];
    const float* W_xproj = (const float*)d_in[6];
    const float* W_dt    = (const float*)d_in[7];
    const float* b_dt    = (const float*)d_in[8];
    const float* A_log   = (const float*)d_in[9];
    const float* D_skip  = (const float*)d_in[10];
    const float* W_out   = (const float*)d_in[11];
    const float* norm_w  = (const float*)d_in[12];
    const float* normf_w = (const float*)d_in[13];
    const float* normf_b = (const float*)d_in[14];
    float* out = (float*)d_out;

    // ---- workspace bump allocator (256B aligned) ----
    char* ws = (char*)d_ws;
    size_t off = 0;
    auto alloc = [&](size_t bytes) -> char* {
        char* p = ws + off;
        off = (off + bytes + 255) & ~((size_t)255);
        return p;
    };
    __bf16* extra_b  = (__bf16*)alloc((size_t)BL * EMBED * 2);
    __bf16* Win_b    = (__bf16*)alloc((size_t)DEPTH * 2 * DIN * EMBED * 2);
    __bf16* Wex_b    = (__bf16*)alloc((size_t)DEPTH * DIN * EMBED * 2);
    __bf16* Wxp_b    = (__bf16*)alloc((size_t)DEPTH * NPROJ * DIN * 2);
    __bf16* Wdt_b    = (__bf16*)alloc((size_t)DEPTH * DIN * 64 * 2);   // K padded 48->64
    __bf16* Wout_b   = (__bf16*)alloc((size_t)DEPTH * EMBED * DIN * 2);
    float*  resid    = (float*)alloc((size_t)BL * EMBED * 4);
    float*  hidden   = (float*)alloc((size_t)BL * EMBED * 4);
    __bf16* hn_b     = (__bf16*)alloc((size_t)BL * EMBED * 2);
    float*  xz       = (float*)alloc((size_t)BL * 2 * DIN * 4);
    float*  u        = (float*)alloc((size_t)BL * DIN * 4);
    __bf16* e_b      = (__bf16*)alloc((size_t)BL * DIN * 2);
    float*  dbl      = (float*)alloc((size_t)BL * NPROJ * 4);
    __bf16* dtlow_b  = (__bf16*)alloc((size_t)BL * 64 * 2);
    float*  dtpre    = (float*)alloc((size_t)BL * DIN * 4);
    __bf16* ymul_b   = (__bf16*)alloc((size_t)BL * DIN * 2);

    auto cvt = [&](const float* s, __bf16* dst, int rows, int ss, int cin, int cout) {
        int total = rows * cout;
        cvt_pad_kernel<<<(total + 255) / 256, 256, 0, stream>>>(s, dst, rows, ss, cin, cout);
    };
    auto gemm = [&](const __bf16* A, const __bf16* B, void* C, int M, int N, int K, int epi) {
        dim3 grid((N + 127) / 128, M / 64);
        if (epi == 0) gemm_nt_bf16<0><<<grid, 256, 0, stream>>>(A, B, C, N, K);
        else          gemm_nt_bf16<1><<<grid, 256, 0, stream>>>(A, B, C, N, K);
    };

    // ---- one-time bf16 conversions ----
    cvt(x_attn,  extra_b, BL,                EMBED,  EMBED,  EMBED);
    cvt(W_in,    Win_b,   DEPTH * 2 * DIN,   EMBED,  EMBED,  EMBED);
    cvt(W_extra, Wex_b,   DEPTH * DIN,       EMBED,  EMBED,  EMBED);
    cvt(W_xproj, Wxp_b,   DEPTH * NPROJ,     DIN,    DIN,    DIN);
    cvt(W_dt,    Wdt_b,   DEPTH * DIN,       DTRANK, DTRANK, 64);
    cvt(W_out,   Wout_b,  DEPTH * EMBED,     DIN,    DIN,    DIN);

    // ---- layer loop ----
    for (int i = 0; i < DEPTH; ++i) {
        const float* h_in = (i == 0) ? x_mamba : hidden;
        rmsnorm_kernel<<<BL, 256, 0, stream>>>(h_in, resid, norm_w + (size_t)i * EMBED, hn_b,
                                               i == 0 ? 1 : 0);

        // xz = hn @ W_in^T  (4096 x 3072, K=768)
        gemm(hn_b, Win_b + (size_t)i * 2 * DIN * EMBED, xz, BL, 2 * DIN, EMBED, 0);

        // u = silu(conv1d(x))
        conv_silu_kernel<<<BL * DIN / 256, 256, 0, stream>>>(
            xz, conv_w + (size_t)i * DIN * DCONV, conv_b + (size_t)i * DIN, u);

        // e = silu(extra @ W_extra^T), bf16 out (fused epilogue)
        gemm(extra_b, Wex_b + (size_t)i * DIN * EMBED, e_b, BL, DIN, EMBED, 1);

        // dbl = e @ W_xproj^T  (4096 x 80, K=1536)
        gemm(e_b, Wxp_b + (size_t)i * NPROJ * DIN, dbl, BL, NPROJ, DIN, 0);

        // dt_low (first 48 cols of dbl) -> bf16, K-padded to 64
        cvt(dbl, dtlow_b, BL, NPROJ, DTRANK, 64);

        // dt_pre = dt_low @ W_dt^T  (4096 x 1536, K=64)
        gemm(dtlow_b, Wdt_b + (size_t)i * DIN * 64, dtpre, BL, DIN, 64, 0);

        // selective scan + D-skip + silu(z) gate -> ymul (bf16)
        scan_kernel<<<dim3(DIN / 256, BATCH), 256, 0, stream>>>(
            dbl, dtpre, u, xz, A_log + (size_t)i * DIN * DSTATE,
            b_dt + (size_t)i * DIN, D_skip + (size_t)i * DIN, ymul_b);

        // hidden = ymul @ W_out^T  (4096 x 768, K=1536)
        gemm(ymul_b, Wout_b + (size_t)i * EMBED * DIN, hidden, BL, EMBED, DIN, 0);
    }

    final_norm_kernel<<<BL, 256, 0, stream>>>(hidden, resid, normf_w, normf_b, out);
}
